// RelationalLayer_5299989643314
// MI455X (gfx1250) — matrix operand order repaired and checked by reading, not compiled, not run on hardware
//
#include <hip/hip_runtime.h>
#include <hip/hip_bf16.h>

typedef __attribute__((ext_vector_type(16))) _Float16 v16h;
typedef __attribute__((ext_vector_type(8)))  _Float16 v8h;
typedef __attribute__((ext_vector_type(8)))  float    v8f;

#define B_SZ 32
#define D_SZ 64
#define K_SZ 256
#define WAVES_PER_WG 4
#define M_PANELS 4            // 4 x 16 = 64 rows per wave (register blocking)

// ---------------------------------------------------------------------------
// Zero x_g accumulator (harness poisons ws; atomics need zeros every call).
__global__ void zero_kernel(float* __restrict__ p, int n) {
    int t = blockIdx.x * blockDim.x + threadIdx.x;
    if (t < n) p[t] = 0.0f;
}

// ---------------------------------------------------------------------------
// u[b,d,:] = x[b,d,:] @ edge_w[:256,:]; v[b,d,:] = x[b,d,:] @ edge_w[256:,:]
__global__ __launch_bounds__(256) void edge_kernel(
    const float* __restrict__ x, const float* __restrict__ ew,
    float* __restrict__ u, float* __restrict__ v) {
    int t   = blockIdx.x * blockDim.x + threadIdx.x;   // 2048*256 threads
    int n   = t & 255;
    int row = t >> 8;
    const float* xr = x + row * 256;
    float au = 0.0f, av = 0.0f;
    for (int k = 0; k < 256; ++k) {
        float xv = xr[k];
        au += xv * ew[k * 256 + n];
        av += xv * ew[(k + 256) * 256 + n];
    }
    u[row * 256 + n] = au;
    v[row * 256 + n] = av;
}

// ---------------------------------------------------------------------------
// Repack g weights [256(k),256(n)] f32 -> f16 WMMA B fragments.
// Per-layer packed index: ((kt*16 + nt)*32 + lane)*16 + e
//   n = nt*16 + (lane & 15)
//   k = kt*32 + (lane>>4)*8 + (e&7) + (e>>3)*16   (B layout: lanes 0-15 K={0..7,16..23}, 16-31 K={8..15,24..31})
__global__ __launch_bounds__(256) void pack_g_weights(
    const float* __restrict__ g0, const float* __restrict__ g1,
    const float* __restrict__ g2, _Float16* __restrict__ wp) {
    int t = blockIdx.x * blockDim.x + threadIdx.x;     // 3*65536 threads
    int l = t >> 16;
    int r = t & 65535;
    const float* gw = (l == 0) ? g0 : ((l == 1) ? g1 : g2);
    int e    = r & 15;
    int lane = (r >> 4) & 31;
    int nt   = (r >> 9) & 15;
    int kt   = r >> 13;
    int n = nt * 16 + (lane & 15);
    int k = kt * 32 + (lane >> 4) * 8 + (e & 7) + (e >> 3) * 16;
    wp[t] = (_Float16)gw[k * 256 + n];
}

// ---------------------------------------------------------------------------
// Main relational kernel: one wave per 64-pair tile (4 register-blocked M
// panels), 3x(256x256) WMMA chain. All 8 B fragments of an nt-tile are
// loaded up front (16 b128 loads in flight) and each feeds 4 WMMAs.
// 64-row tile == one full i-row (j = 0..63), so v[b,i,:] is loop-invariant.
__global__ __launch_bounds__(WAVES_PER_WG * 32) void g_main(
    const float* __restrict__ u, const float* __restrict__ v,
    const float* __restrict__ eb, const _Float16* __restrict__ wpack,
    const float* __restrict__ gb0, const float* __restrict__ gb1,
    const float* __restrict__ gb2, float* __restrict__ xg) {
    extern __shared__ _Float16 lds[];                   // WAVES*64*256 f16

    int wave = threadIdx.x >> 5;
    int lane = threadIdx.x & 31;
    _Float16* h = lds + wave * 64 * 256;                // 32 KB per wave

    int mtile = blockIdx.x * WAVES_PER_WG + wave;       // 0..2047
    int b     = mtile >> 6;                             // 64 tiles / batch
    int i_idx = mtile & 63;                             // tile == one i row

    int m     = lane & 15;
    int khalf = lane >> 4;

    // ---- build h[j,:] = u[b,j,:] + v[b,i,:] + eb  (f16 into LDS tile) -----
    {
        int c = lane * 8;                               // 32 lanes * 8 = 256
        const float4* ep = (const float4*)(eb + c);
        float4 e0 = ep[0], e1 = ep[1];
        const float4* vp = (const float4*)(v + ((b * D_SZ + i_idx) * 256 + c));
        float4 v0 = vp[0], v1 = vp[1];
        for (int j = 0; j < 64; ++j) {
            const float4* up = (const float4*)(u + ((b * D_SZ + j) * 256 + c));
            float4 u0 = up[0], u1 = up[1];
            v8h hv;
            hv[0] = (_Float16)(u0.x + v0.x + e0.x);
            hv[1] = (_Float16)(u0.y + v0.y + e0.y);
            hv[2] = (_Float16)(u0.z + v0.z + e0.z);
            hv[3] = (_Float16)(u0.w + v0.w + e0.w);
            hv[4] = (_Float16)(u1.x + v1.x + e1.x);
            hv[5] = (_Float16)(u1.y + v1.y + e1.y);
            hv[6] = (_Float16)(u1.z + v1.z + e1.z);
            hv[7] = (_Float16)(u1.w + v1.w + e1.w);
            *(v8h*)(h + j * 256 + c) = hv;
        }
    }
    asm volatile("s_wait_dscnt 0" ::: "memory");

    // ---- 3 g-layers ------------------------------------------------------
    for (int l = 0; l < 3; ++l) {
        // Load 4 A panels (each 16x256 f16) from LDS, ISA A layout:
        // lane group khalf: k_local = {0..7,16..23} + khalf*8
        v16h A[M_PANELS][8];
#pragma unroll
        for (int mp = 0; mp < M_PANELS; ++mp) {
            int row = mp * 16 + m;
#pragma unroll
            for (int kt = 0; kt < 8; ++kt) {
                v8h lo = *(const v8h*)(h + row * 256 + kt * 32 + khalf * 8);
                v8h hi = *(const v8h*)(h + row * 256 + kt * 32 + 16 + khalf * 8);
#pragma unroll
                for (int e = 0; e < 8; ++e) {
                    A[mp][kt][e] = lo[e];
                    A[mp][kt][8 + e] = hi[e];
                }
            }
        }
        asm volatile("s_wait_dscnt 0" ::: "memory");

        const _Float16* wl = wpack + l * 65536;
        const float*    gb = (l == 0) ? gb0 : ((l == 1) ? gb1 : gb2);

        for (int nt = 0; nt < 16; ++nt) {
            // Prefetch all 8 B fragments of this nt-tile (16 b128 in flight)
            v16h Bfrag[8];
#pragma unroll
            for (int kt = 0; kt < 8; ++kt)
                Bfrag[kt] =
                    *(const v16h*)(wl + (((kt * 16 + nt) * 32) + lane) * 16);

            float bv = gb[nt * 16 + m];
            v8f acc[M_PANELS];
#pragma unroll
            for (int mp = 0; mp < M_PANELS; ++mp)
#pragma unroll
                for (int r = 0; r < 8; ++r) acc[mp][r] = bv;

#pragma unroll
            for (int kt = 0; kt < 8; ++kt) {
#pragma unroll
                for (int mp = 0; mp < M_PANELS; ++mp) {
                    acc[mp] = __builtin_amdgcn_wmma_f32_16x16x32_f16(
                        false, A[mp][kt], false, Bfrag[kt], (short)0, acc[mp],
                        false, false);
                }
            }
#pragma unroll
            for (int mp = 0; mp < M_PANELS; ++mp)
#pragma unroll
                for (int r = 0; r < 8; ++r)
                    acc[mp][r] = fmaxf(acc[mp][r], 0.0f);

            if (l < 2) {
                // C layout: lane holds n = nt*16+m, rows r + khalf*8 (+mp*16)
                int n = nt * 16 + m;
#pragma unroll
                for (int mp = 0; mp < M_PANELS; ++mp)
#pragma unroll
                    for (int r = 0; r < 8; ++r)
                        h[(mp * 16 + r + khalf * 8) * 256 + n] =
                            (_Float16)acc[mp][r];
            } else {
                // reduce over all 64 rows of this tile, accumulate into x_g
                float s = 0.0f;
#pragma unroll
                for (int mp = 0; mp < M_PANELS; ++mp)
#pragma unroll
                    for (int r = 0; r < 8; ++r) s += acc[mp][r];
                s += __shfl_xor(s, 16, 32);
                if (lane < 16) atomicAdd(&xg[b * 256 + nt * 16 + lane], s);
            }
        }
        asm volatile("s_wait_dscnt 0" ::: "memory");
    }
}

// ---------------------------------------------------------------------------
// f-MLP: x_g[32,256] -> relu(f1) -> relu(f2) -> f3 -> out[32,128]. Tiny.
__global__ __launch_bounds__(256) void f_kernel(
    const float* __restrict__ xg,
    const float* __restrict__ f1w, const float* __restrict__ f1b,
    const float* __restrict__ f2w, const float* __restrict__ f2b,
    const float* __restrict__ f3w, const float* __restrict__ f3b,
    float* __restrict__ out) {
    __shared__ float s0[256], s1[256], s2[256];
    int t = threadIdx.x;
    for (int b = 0; b < B_SZ; ++b) {
        s0[t] = xg[b * 256 + t];
        __syncthreads();
        float a = f1b[t];
        for (int k = 0; k < 256; ++k) a += s0[k] * f1w[k * 256 + t];
        s1[t] = fmaxf(a, 0.0f);
        __syncthreads();
        a = f2b[t];
        for (int k = 0; k < 256; ++k) a += s1[k] * f2w[k * 256 + t];
        s2[t] = fmaxf(a, 0.0f);
        __syncthreads();
        if (t < 128) {
            a = f3b[t];
            for (int k = 0; k < 256; ++k) a += s2[k] * f3w[k * 128 + t];
            out[b * 128 + t] = a;
        }
        __syncthreads();
    }
}

// ---------------------------------------------------------------------------
extern "C" void kernel_launch(void* const* d_in, const int* in_sizes, int n_in,
                              void* d_out, int out_size, void* d_ws, size_t ws_size,
                              hipStream_t stream) {
    const float* x      = (const float*)d_in[0];
    const float* edge_w = (const float*)d_in[1];
    const float* edge_b = (const float*)d_in[2];
    const float* gw0    = (const float*)d_in[3];
    const float* gw1    = (const float*)d_in[4];
    const float* gw2    = (const float*)d_in[5];
    const float* gb0    = (const float*)d_in[6];
    const float* gb1    = (const float*)d_in[7];
    const float* gb2    = (const float*)d_in[8];
    const float* f1w    = (const float*)d_in[9];
    const float* f1b    = (const float*)d_in[10];
    const float* f2w    = (const float*)d_in[11];
    const float* f2b    = (const float*)d_in[12];
    const float* f3w    = (const float*)d_in[13];
    const float* f3b    = (const float*)d_in[14];

    char* ws = (char*)d_ws;
    float*    u     = (float*)(ws);                     // 2 MB
    float*    v     = (float*)(ws + 2097152);           // 2 MB
    _Float16* wpack = (_Float16*)(ws + 4194304);        // 384 KB
    float*    xg    = (float*)(ws + 4587520);           // 32 KB

    zero_kernel<<<32, 256, 0, stream>>>(xg, B_SZ * 256);
    edge_kernel<<<2048, 256, 0, stream>>>(x, edge_w, u, v);
    pack_g_weights<<<768, 256, 0, stream>>>(gw0, gw1, gw2, wpack);

    // 2048 64-row tiles / 4 waves per WG; dynamic LDS: 4 waves * 32 KB
    size_t lds_bytes = (size_t)WAVES_PER_WG * 64 * 256 * sizeof(_Float16);
    g_main<<<512, WAVES_PER_WG * 32, lds_bytes, stream>>>(
        u, v, edge_b, wpack, gb0, gb1, gb2, xg);

    f_kernel<<<1, 256, 0, stream>>>(xg, f1w, f1b, f2w, f2b, f3w, f3b,
                                    (float*)d_out);
}
